// GlobalAttention_40218073759982
// MI455X (gfx1250) — matrix operand order repaired and checked by reading, not compiled, not run on hardware
//
#include <hip/hip_runtime.h>

// ---------------------------------------------------------------------------
// GlobalAttention (AlphaFold-style) for gfx1250: bf16 WMMA, f32 accumulate,
// async global->LDS staging for the attention-weighted sum.
// B=1, N=384, S=1024, C_IN=256, C_HID=32, H=8.
// ---------------------------------------------------------------------------

#define C_IN    256
#define C_HID   32
#define N_HEADS 8
#define S_LEN   1024
#define N_ROWS  384
#define S_TILE  64
#define V_CHUNK 128   // rows of v staged per async chunk (16 KB fp32)

typedef __attribute__((ext_vector_type(16))) __bf16       v16bf;
typedef __attribute__((ext_vector_type(8)))  float        v8f;
typedef __attribute__((ext_vector_type(8)))  unsigned int v8u;

__device__ __forceinline__ v8f wmma_bf16(v16bf a, v16bf b, v8f c) {
  // emits v_wmma_f32_16x16x32_bf16
  return __builtin_amdgcn_wmma_f32_16x16x32_bf16(
      /*neg_a=*/false, a, /*neg_b=*/false, b,
      /*c_mod=*/(short)0, c, /*reuse_a=*/false, /*reuse_b=*/false);
}

// float -> bf16 with round-to-nearest-even (storage-safe, no fp-conv reliance)
__device__ __forceinline__ unsigned short f2bf_bits(float f) {
  unsigned u = __builtin_bit_cast(unsigned, f);
  u += 0x7FFFu + ((u >> 16) & 1u);
  return (unsigned short)(u >> 16);
}
__device__ __forceinline__ __bf16 f2bf(float f) {
  unsigned short s = f2bf_bits(f);
  return __builtin_bit_cast(__bf16, s);
}
__device__ __forceinline__ unsigned pack2bf(float lo, float hi) {
  return (unsigned)f2bf_bits(lo) | ((unsigned)f2bf_bits(hi) << 16);
}

// A-matrix (16x32 bf16) K-offset for VGPR pair r, element e, lane (contiguous: k = 16*(lane/16) + 2r + e)
__device__ __forceinline__ int a_k_off(int r, int e, int lane) {
  return ((lane >> 4) << 4) | (r << 1) | e;
}

// CDNA5 async copy: LDS[ldsoff..+15] = MEM[gaddr..+15], tracked by ASYNCcnt.
__device__ __forceinline__ void async_cp16(unsigned ldsoff, const void* gaddr) {
  asm volatile("global_load_async_to_lds_b128 %0, %1, off"
               :: "v"(ldsoff), "v"(gaddr) : "memory");
}

// ---------------------------------------------------------------------------
// Kernel 0: repack weights into WMMA-B-fragment-major bf16 layout.
// Fragment f = kc*(nOut/16)+nt holds B[kc*32 .. +31][nt*16 .. +15];
// element (lane, j): col = nt*16+(lane&15), k = kc*32 + ((lane>>4)<<4) + j.
// ---------------------------------------------------------------------------
__global__ void pack_w(const float* __restrict__ Wk, const float* __restrict__ Wv,
                       const float* __restrict__ Wg, const float* __restrict__ Wo,
                       __bf16* fWk, __bf16* fWv, __bf16* fWg, __bf16* fWo) {
  int idx = blockIdx.x * 256 + threadIdx.x;
  const float* W; __bf16* D; int nOut; int local;
  if (idx < 8192)        { W = Wk; D = fWk; nOut = 32;  local = idx; }
  else if (idx < 16384)  { W = Wv; D = fWv; nOut = 32;  local = idx - 8192; }
  else if (idx < 81920)  { W = Wg; D = fWg; nOut = 256; local = idx - 16384; }
  else if (idx < 147456) { W = Wo; D = fWo; nOut = 256; local = idx - 81920; }
  else return;
  int frag = local >> 9, within = local & 511;
  int lane = within >> 4, j = within & 15;
  int npk = nOut >> 4;
  int kc = frag / npk, nt = frag - kc * npk;
  int col = nt * 16 + (lane & 15);
  int k   = kc * 32 + ((lane >> 4) << 4) + j;
  D[local] = f2bf(W[k * nOut + col]);
}

// ---------------------------------------------------------------------------
// Kernel 1: masked mean over S  ->  q_ws[n][256]
// ---------------------------------------------------------------------------
__global__ void qmean_k(const float* __restrict__ m, const float* __restrict__ mask,
                        float* __restrict__ q_ws) {
  int n = blockIdx.x, tid = threadIdx.x;
  __shared__ float smask[S_LEN];
  __shared__ float red[256];
  float part = 0.f;
  for (int s = tid; s < S_LEN; s += 256) {
    float mv = mask[n * S_LEN + s];
    smask[s] = mv; part += mv;
  }
  red[tid] = part; __syncthreads();
  for (int off = 128; off >= 1; off >>= 1) {
    if (tid < off) red[tid] += red[tid + off];
    __syncthreads();
  }
  float denom = red[0] + 1e-10f;
  float acc = 0.f;
  const float* base = m + (size_t)n * S_LEN * C_IN + tid;
  for (int s = 0; s < S_LEN; ++s) acc += base[s * C_IN] * smask[s];
  q_ws[n * C_IN + tid] = acc / denom;
}

// ---------------------------------------------------------------------------
// Kernel 2: qh = (q @ Wq) * C_HID^-0.5   ->  qh_ws[n][256]  ([h*32+c])
// ---------------------------------------------------------------------------
__global__ void qproj_k(const float* __restrict__ q_ws, const float* __restrict__ Wq,
                        float* __restrict__ qh_ws) {
  int n = blockIdx.x, j = threadIdx.x;
  __shared__ float sq[C_IN];
  sq[j] = q_ws[n * C_IN + j];
  __syncthreads();
  float acc = 0.f;
#pragma unroll 8
  for (int c = 0; c < C_IN; ++c) acc += sq[c] * Wq[c * C_IN + j];
  qh_ws[n * C_IN + j] = acc * 0.17677669529663687f; // 32^-0.5
}

// ---------------------------------------------------------------------------
// Kernel 3: per 64-row tile: k = m@Wk, v = m@Wv (WMMA), logits a = q.k^T (WMMA)
// ---------------------------------------------------------------------------
__global__ void kv_logits_k(const float* __restrict__ m, const float* __restrict__ qh_ws,
                            const __bf16* __restrict__ fWk, const __bf16* __restrict__ fWv,
                            float* __restrict__ a_ws, float* __restrict__ v_ws) {
  int n = blockIdx.y, st = blockIdx.x * S_TILE;
  int tid = threadIdx.x, lane = tid & 31, wave = tid >> 5;
  __shared__ __bf16 lm[S_TILE * C_IN];   // 32 KB: m tile in bf16
  __shared__ __bf16 kt[S_TILE * C_HID];  //  4 KB: k tile in bf16

  // stage m tile -> LDS (paired bf16 stores)
  const float* msrc = m + (size_t)(n * S_LEN + st) * C_IN;
#pragma unroll
  for (int i = 0; i < (S_TILE * C_IN) / 512; ++i) {
    float2 v2 = ((const float2*)msrc)[i * 256 + tid];
    *(unsigned*)&lm[(i * 256 + tid) * 2] = pack2bf(v2.x, v2.y);
  }
  __syncthreads();

  int mt = wave & 3;                       // M-tile (16 rows of s)
  const __bf16* wf = (wave < 4) ? fWk : fWv;
  int mrow = mt * 16 + (lane & 15);
  int rl   = (lane >> 4) << 3;

  v8f acc0 = {}, acc1 = {};
  for (int kc = 0; kc < 8; ++kc) {
    v8u au;
#pragma unroll
    for (int r = 0; r < 8; ++r)
      au[r] = *(const unsigned*)&lm[mrow * C_IN + kc * 32 + a_k_off(r, 0, lane)];
    v16bf A = __builtin_bit_cast(v16bf, au);
#pragma unroll
    for (int nt = 0; nt < 2; ++nt) {
      v8u  braw = *(const v8u*)(wf + (((kc * 2) + nt) << 9) + lane * 16);
      v16bf B   = __builtin_bit_cast(v16bf, braw);
      if (nt == 0) acc0 = wmma_bf16(A, B, acc0);
      else         acc1 = wmma_bf16(A, B, acc1);
    }
  }

  if (wave >= 4) { // v waves -> global ws (fp32)
#pragma unroll
    for (int nt = 0; nt < 2; ++nt) {
      v8f a = nt ? acc1 : acc0;
#pragma unroll
      for (int r = 0; r < 8; ++r) {
        int row = mt * 16 + rl + r, col = nt * 16 + (lane & 15);
        v_ws[(size_t)(n * S_LEN + st + row) * C_HID + col] = a[r];
      }
    }
  } else {         // k waves -> LDS (bf16) for logits
#pragma unroll
    for (int nt = 0; nt < 2; ++nt) {
      v8f a = nt ? acc1 : acc0;
#pragma unroll
      for (int r = 0; r < 8; ++r) {
        int row = mt * 16 + rl + r, col = nt * 16 + (lane & 15);
        kt[row * C_HID + col] = f2bf(a[r]);
      }
    }
  }
  __syncthreads();

  if (wave < 4) {  // logits: D[h][s16] = qh(16x32, heads padded) x k^T(32x16)
    int h = lane & 15;
    v8u qau;
#pragma unroll
    for (int r = 0; r < 8; ++r) {
      unsigned pr = 0;
      if (h < N_HEADS) {
        float2 q2 = *(const float2*)&qh_ws[n * C_IN + h * C_HID + a_k_off(r, 0, lane)];
        pr = pack2bf(q2.x, q2.y);
      }
      qau[r] = pr;
    }
    v16bf Aq = __builtin_bit_cast(v16bf, qau);
    int sl = wave * 16 + (lane & 15);
    int cb = (lane >> 4) << 4;
    v8u  kraw = *(const v8u*)&kt[sl * C_HID + cb];
    v16bf Bk  = __builtin_bit_cast(v16bf, kraw);
    v8f accA = {};
    accA = wmma_bf16(Aq, Bk, accA);
    if (lane < 16) { // rows 0..7 (= heads 0..7) live in lanes 0-15
#pragma unroll
      for (int r = 0; r < 8; ++r)
        a_ws[(size_t)(n * N_HEADS + r) * S_LEN + st + wave * 16 + lane] = accA[r];
    }
  }
}

// ---------------------------------------------------------------------------
// Kernel 4: mask bias + softmax over S, then o[h][c] = sum_s p*v.
// v is staged into LDS in double-buffered 128-row chunks via the CDNA5
// async global->LDS path (ASYNCcnt), so each v element is fetched once per
// block and all 8 head-waves consume it from LDS.
// ---------------------------------------------------------------------------
__global__ void softmax_o_k(const float* __restrict__ a_ws, const float* __restrict__ mask,
                            const float* __restrict__ v_ws, float* __restrict__ o_ws) {
  int n = blockIdx.x, tid = threadIdx.x, h = tid >> 5, l = tid & 31;
  __shared__ float p[N_HEADS * S_LEN];              // 32 KB
  __shared__ float sv[2][V_CHUNK * C_HID];          // 2 x 16 KB v staging
  __shared__ float red[256];
  const float* arow = a_ws + (size_t)(n * N_HEADS + h) * S_LEN;
  const float* mrow = mask + n * S_LEN;
  const float* vbase = v_ws + (size_t)n * S_LEN * C_HID;

  // kick off async copy of v chunk 0 (each thread moves 4x16B, in-order)
  {
#pragma unroll
    for (int i = 0; i < (V_CHUNK * C_HID) / (4 * 256); ++i) {
      int f16 = i * 256 + tid; // which 16-byte packet
      async_cp16((unsigned)(size_t)&sv[0][f16 * 4], vbase + f16 * 4);
    }
  }

  float mx = -3.4e38f;
  for (int s = l; s < S_LEN; s += 32) {
    float v = arow[s] + 1e9f * (mrow[s] - 1.0f);
    p[h * S_LEN + s] = v;
    mx = fmaxf(mx, v);
  }
  red[tid] = mx; __syncthreads();
  for (int off = 16; off >= 1; off >>= 1) {
    if (l < off) red[tid] = fmaxf(red[tid], red[tid + off]);
    __syncthreads();
  }
  mx = red[h * 32];
  __syncthreads();

  float se = 0.f;
  for (int s = l; s < S_LEN; s += 32) {
    float ex = __expf(p[h * S_LEN + s] - mx);
    p[h * S_LEN + s] = ex;
    se += ex;
  }
  red[tid] = se; __syncthreads();
  for (int off = 16; off >= 1; off >>= 1) {
    if (l < off) red[tid] += red[tid + off];
    __syncthreads();
  }
  float inv = 1.0f / red[h * 32];
  __syncthreads();

  // o[h][c] accumulation over double-buffered v chunks
  float o = 0.f;
  const int NCH = S_LEN / V_CHUNK;
  for (int c = 0; c < NCH; ++c) {
    int buf = c & 1;
    if (c + 1 < NCH) { // prefetch next chunk into the other buffer
#pragma unroll
      for (int i = 0; i < (V_CHUNK * C_HID) / (4 * 256); ++i) {
        int f16 = i * 256 + tid;
        async_cp16((unsigned)(size_t)&sv[(c + 1) & 1][f16 * 4],
                   vbase + (size_t)(c + 1) * V_CHUNK * C_HID + f16 * 4);
      }
      // 4 transfers pending for chunk c+1; in-order => chunk c complete
      asm volatile("s_wait_asynccnt 0x4" ::: "memory");
    } else {
      asm volatile("s_wait_asynccnt 0x0" ::: "memory");
    }
    __syncthreads(); // chunk c visible to all waves; prev buffer free to fill
    const float* pv = &p[h * S_LEN + c * V_CHUNK];
#pragma unroll 4
    for (int sl = 0; sl < V_CHUNK; ++sl)
      o += pv[sl] * sv[buf][sl * C_HID + l];
    __syncthreads(); // all waves done reading buf before it is overwritten
  }
  o_ws[n * C_IN + h * C_HID + l] = o * inv;
}

// ---------------------------------------------------------------------------
// Kernel 5: fused g = sigmoid(m@Wg+bg)*o ; out = g@Wo + bo   (128 WMMAs/wave)
// ---------------------------------------------------------------------------
__global__ void gate_out_k(const float* __restrict__ m, const float* __restrict__ o_ws,
                           const float* __restrict__ bg, const float* __restrict__ bo,
                           const __bf16* __restrict__ fWg, const __bf16* __restrict__ fWo,
                           float* __restrict__ out) {
  int n = blockIdx.y, st = blockIdx.x * S_TILE;
  int tid = threadIdx.x, lane = tid & 31, wave = tid >> 5;
  __shared__ __bf16 lt[S_TILE * C_IN];  // 32 KB, reused m-tile then g-tile

  const float* msrc = m + (size_t)(n * S_LEN + st) * C_IN;
#pragma unroll
  for (int i = 0; i < (S_TILE * C_IN) / 512; ++i) {
    float2 v2 = ((const float2*)msrc)[i * 256 + tid];
    *(unsigned*)&lt[(i * 256 + tid) * 2] = pack2bf(v2.x, v2.y);
  }
  __syncthreads();

  int mt = wave >> 1, nb = (wave & 1) * 8; // M-tile, N-tile base (8 tiles wide)
  int mrow = mt * 16 + (lane & 15);
  int rl   = (lane >> 4) << 3;

  // ---- GEMM1: t = m @ Wg ----
  v8f acc[8] = {};
  for (int kc = 0; kc < 8; ++kc) {
    v8u au;
#pragma unroll
    for (int r = 0; r < 8; ++r)
      au[r] = *(const unsigned*)&lt[mrow * C_IN + kc * 32 + a_k_off(r, 0, lane)];
    v16bf A = __builtin_bit_cast(v16bf, au);
#pragma unroll
    for (int nt = 0; nt < 8; ++nt) {
      v8u  braw = *(const v8u*)(fWg + ((kc * 16 + nb + nt) << 9) + lane * 16);
      v16bf B   = __builtin_bit_cast(v16bf, braw);
      acc[nt] = wmma_bf16(A, B, acc[nt]);
    }
  }
  __syncthreads(); // everyone done reading m-tile

  // ---- epilogue: g = sigmoid(t + bg) * o[n], back to LDS as bf16 ----
#pragma unroll
  for (int nt = 0; nt < 8; ++nt) {
    int col = (nb + nt) * 16 + (lane & 15);
    float bgv = bg[col];
    float ov  = o_ws[n * C_IN + col];
#pragma unroll
    for (int r = 0; r < 8; ++r) {
      int row = mt * 16 + rl + r;
      float x = acc[nt][r] + bgv;
      float g = ov / (1.0f + __expf(-x));
      lt[row * C_IN + col] = f2bf(g);
    }
  }
  __syncthreads();

  // ---- GEMM2: out = g @ Wo + bo ----
  v8f acc2[8] = {};
  for (int kc = 0; kc < 8; ++kc) {
    v8u au;
#pragma unroll
    for (int r = 0; r < 8; ++r)
      au[r] = *(const unsigned*)&lt[mrow * C_IN + kc * 32 + a_k_off(r, 0, lane)];
    v16bf A = __builtin_bit_cast(v16bf, au);
#pragma unroll
    for (int nt = 0; nt < 8; ++nt) {
      v8u  braw = *(const v8u*)(fWo + ((kc * 16 + nb + nt) << 9) + lane * 16);
      v16bf B   = __builtin_bit_cast(v16bf, braw);
      acc2[nt] = wmma_bf16(A, B, acc2[nt]);
    }
  }
#pragma unroll
  for (int nt = 0; nt < 8; ++nt) {
    int col = (nb + nt) * 16 + (lane & 15);
    float bov = bo[col];
#pragma unroll
    for (int r = 0; r < 8; ++r) {
      int row = mt * 16 + rl + r;
      out[(size_t)(n * S_LEN + st + row) * C_IN + col] = acc2[nt][r] + bov;
    }
  }
}

// ---------------------------------------------------------------------------
extern "C" void kernel_launch(void* const* d_in, const int* in_sizes, int n_in,
                              void* d_out, int out_size, void* d_ws, size_t ws_size,
                              hipStream_t stream) {
  const float* m    = (const float*)d_in[0];
  const float* mask = (const float*)d_in[1];
  const float* Wq   = (const float*)d_in[2];
  const float* Wk   = (const float*)d_in[3];
  const float* Wv   = (const float*)d_in[4];
  const float* Wg   = (const float*)d_in[5];
  const float* bg   = (const float*)d_in[6];
  const float* Wo   = (const float*)d_in[7];
  const float* bo   = (const float*)d_in[8];
  float* out = (float*)d_out;

  size_t off = 0;
  auto alloc = [&](size_t bytes) {
    void* p = (char*)d_ws + off;
    off += (bytes + 255) & ~(size_t)255;
    return p;
  };
  float*  q_ws  = (float*)alloc((size_t)N_ROWS * C_IN * 4);
  float*  qh_ws = (float*)alloc((size_t)N_ROWS * C_IN * 4);
  float*  a_ws  = (float*)alloc((size_t)N_ROWS * N_HEADS * S_LEN * 4);
  float*  v_ws  = (float*)alloc((size_t)N_ROWS * S_LEN * C_HID * 4);
  float*  o_ws  = (float*)alloc((size_t)N_ROWS * C_IN * 4);
  __bf16* fWk   = (__bf16*)alloc((size_t)16 * 512 * 2);
  __bf16* fWv   = (__bf16*)alloc((size_t)16 * 512 * 2);
  __bf16* fWg   = (__bf16*)alloc((size_t)128 * 512 * 2);
  __bf16* fWo   = (__bf16*)alloc((size_t)128 * 512 * 2);

  pack_w<<<576, 256, 0, stream>>>(Wk, Wv, Wg, Wo, fWk, fWv, fWg, fWo);
  qmean_k<<<N_ROWS, 256, 0, stream>>>(m, mask, q_ws);
  qproj_k<<<N_ROWS, 256, 0, stream>>>(q_ws, Wq, qh_ws);
  kv_logits_k<<<dim3(S_LEN / S_TILE, N_ROWS), 256, 0, stream>>>(m, qh_ws, fWk, fWv, a_ws, v_ws);
  softmax_o_k<<<N_ROWS, 256, 0, stream>>>(a_ws, mask, v_ws, o_ws);
  gate_out_k<<<dim3(S_LEN / S_TILE, N_ROWS), 256, 0, stream>>>(m, o_ws, bg, bo, fWg, fWo, out);
}
